// Gate_8108898255611
// MI455X (gfx1250) — matrix operand order repaired and checked by reading, not compiled, not run on hardware
//
#include <hip/hip_runtime.h>
#include <hip/hip_bf16.h>

// ---------------------------------------------------------------------------
// MoE router gate for MI455X (gfx1250, wave32, WMMA)
//   scores = x[16384,4096] @ weight[256,4096]^T  (bf16 WMMA, f32 accum)
//   fused softmax->argmax:  out_w[t] = 1/sum(exp(s - max)),  out_i[t] = argmax
// ---------------------------------------------------------------------------

#define NUM_TOKENS 16384
#define DIM        4096
#define N_EXPERTS  256
#define ROUTE_SCALE 1.0f

typedef __attribute__((ext_vector_type(16))) __bf16 v16bf;
typedef __attribute__((ext_vector_type(8)))  float  v8f;

constexpr int BM      = 32;              // tokens per workgroup
constexpr int BK      = 32;              // K per WMMA step (16x16x32 bf16)
constexpr int NWAVE   = 16;              // one 16-expert tile per wave -> 256 experts
constexpr int THREADS = NWAVE * 32;
constexpr int PAD_BF  = 8;               // bf16 pad to break LDS bank alignment
constexpr int AROW    = DIM + PAD_BF;    // bf16 elements per LDS A row
constexpr int AROWB   = AROW * 2;        // bytes per LDS A row (8208)
constexpr int SROW    = N_EXPERTS + 1;   // score row stride (floats), padded
constexpr size_t SMEM_BYTES = (size_t)BM * AROWB + (size_t)BM * SROW * 4;
static_assert(SMEM_BYTES <= 320u * 1024u, "exceeds WGP LDS");

// f32 -> bf16 round-to-nearest-even, two at a time, packed into a dword
__device__ __forceinline__ unsigned int pack_bf2(float a, float b) {
    unsigned int ua = __float_as_uint(a);
    unsigned int ub = __float_as_uint(b);
    ua = (ua + 0x7FFFu + ((ua >> 16) & 1u)) >> 16;
    ub = (ub + 0x7FFFu + ((ub >> 16) & 1u)) >> 16;
    return ua | (ub << 16);
}

// Pre-convert weight [256,4096] f32 -> packed bf16 in workspace (2 MB).
__global__ void cvt_weight_kernel(const float* __restrict__ w,
                                  unsigned int* __restrict__ wb) {
    int i = blockIdx.x * blockDim.x + threadIdx.x;   // pair index
    const float2 f = ((const float2*)w)[i];
    wb[i] = pack_bf2(f.x, f.y);
}

template <bool PRECONV>
__global__ __launch_bounds__(THREADS)
void router_kernel(const float* __restrict__ x,
                   const float* __restrict__ wgt,
                   const unsigned short* __restrict__ wgt_bf,
                   float* __restrict__ out) {
    extern __shared__ char smem[];
    char*  aLds   = smem;                                   // [BM][AROW] bf16
    float* scores = (float*)(smem + (size_t)BM * AROWB);    // [BM][SROW] f32

    const int tid  = threadIdx.x;
    const int wave = tid >> 5;
    const int lane = tid & 31;
    const int l16  = lane & 15;
    const int half = lane >> 4;
    const int tok0 = blockIdx.x * BM;

    // ---- stage x block into LDS as bf16 (coalesced b128 loads) ----
    {
        const float4* xv = (const float4*)(x + (size_t)tok0 * DIM);
        for (int q = tid; q < BM * (DIM / 4); q += THREADS) {
            int row = q >> 10;            // DIM/4 == 1024 quads per row
            int c4  = q & 1023;
            float4 f = xv[(size_t)row * (DIM / 4) + c4];
            uint2 p;
            p.x = pack_bf2(f.x, f.y);
            p.y = pack_bf2(f.z, f.w);
            *(uint2*)(aLds + (size_t)row * AROWB + (size_t)c4 * 8) = p;
        }
    }
    __syncthreads();

    // ---- K loop: each wave does one 16-expert N-tile x two 16-token M-tiles
    v8f acc0 = {};
    v8f acc1 = {};
    const int expert = wave * 16 + l16;                  // B row this lane feeds
    // A operand in this code's contiguous K order (not the ISA table order): lane holds row M=l16; VGPR0..7 = K 16h..16h+15
    // (same K order as B)  -> two contiguous 16B LDS reads per tile.
    const char* aBase0 = aLds + (size_t)l16 * AROWB + half * 32;
    const char* aBase1 = aLds + (size_t)(l16 + 16) * AROWB + half * 32;

#pragma unroll 2
    for (int kk = 0; kk < DIM / BK; ++kk) {
        const int kb = kk * BK;

        union { uint4 u[2]; v16bf v; } a0, a1;
        a0.u[0] = *(const uint4*)(aBase0 + (size_t)kb * 2);
        a0.u[1] = *(const uint4*)(aBase0 + (size_t)kb * 2 + 16);
        a1.u[0] = *(const uint4*)(aBase1 + (size_t)kb * 2);
        a1.u[1] = *(const uint4*)(aBase1 + (size_t)kb * 2 + 16);

        // B operand in this code's contiguous K order (matches A): lane (N=l16, h) holds K = 16h..16h+15
        // packed pairwise in VGPR0..7 -> 32 contiguous bytes per lane.
        union { uint4 u[2]; unsigned int w[8]; v16bf v; } b;
        if (PRECONV) {
            const char* bp = (const char*)wgt_bf +
                             ((size_t)expert * DIM + kb + half * 16) * 2;
            b.u[0] = *(const uint4*)bp;
            b.u[1] = *(const uint4*)(bp + 16);
        } else {
            const float* bp = wgt + (size_t)expert * DIM + kb + half * 16;
            float4 f0 = *(const float4*)(bp + 0);
            float4 f1 = *(const float4*)(bp + 4);
            float4 f2 = *(const float4*)(bp + 8);
            float4 f3 = *(const float4*)(bp + 12);
            b.w[0] = pack_bf2(f0.x, f0.y); b.w[1] = pack_bf2(f0.z, f0.w);
            b.w[2] = pack_bf2(f1.x, f1.y); b.w[3] = pack_bf2(f1.z, f1.w);
            b.w[4] = pack_bf2(f2.x, f2.y); b.w[5] = pack_bf2(f2.z, f2.w);
            b.w[6] = pack_bf2(f3.x, f3.y); b.w[7] = pack_bf2(f3.z, f3.w);
        }

        acc0 = __builtin_amdgcn_wmma_f32_16x16x32_bf16(
                   false, a0.v, false, b.v, (short)0, acc0, false, false);
        acc1 = __builtin_amdgcn_wmma_f32_16x16x32_bf16(
                   false, a1.v, false, b.v, (short)0, acc1, false, false);
    }

    // ---- scatter C tiles to LDS scores[token][expert] ----
    // C/D layout: lanes 0-15 -> N=lane, VGPR g -> M=g; lanes 16-31 -> M=g+8.
    {
        const int col = wave * 16 + l16;
#pragma unroll
        for (int g = 0; g < 8; ++g) {
            scores[(g + 8 * half) * SROW + col]        = acc0[g];
            scores[(16 + g + 8 * half) * SROW + col]   = acc1[g];
        }
    }
    __syncthreads();

    // ---- fused softmax-argmax: wave w handles tokens w and w+16 ----
    for (int t = wave; t < BM; t += NWAVE) {
        const float* srow = scores + (size_t)t * SROW;
        float sv[8];
        float m   = -3.402823466e38f;
        int   idx = 0;
#pragma unroll
        for (int j = 0; j < 8; ++j) {
            float s = srow[lane + 32 * j];
            sv[j] = s;
            if (s > m) { m = s; idx = lane + 32 * j; }
        }
#pragma unroll
        for (int off = 16; off; off >>= 1) {       // wave32 max+argmax reduce
            float om = __shfl_xor(m, off, 32);
            int   oi = __shfl_xor(idx, off, 32);
            if (om > m || (om == m && oi < idx)) { m = om; idx = oi; }
        }
        float se = 0.0f;
#pragma unroll
        for (int j = 0; j < 8; ++j) se += __expf(sv[j] - m);
#pragma unroll
        for (int off = 16; off; off >>= 1) se += __shfl_xor(se, off, 32);

        if (lane == 0) {
            const int tg = tok0 + t;
            out[tg] = ROUTE_SCALE * (1.0f / se);     // prob at argmax
            ((int*)out)[NUM_TOKENS + tg] = idx;      // argmax index
        }
    }
}

extern "C" void kernel_launch(void* const* d_in, const int* in_sizes, int n_in,
                              void* d_out, int out_size, void* d_ws, size_t ws_size,
                              hipStream_t stream) {
    const float* x = (const float*)d_in[0];   // [16384, 4096] f32
    const float* w = (const float*)d_in[1];   // [256, 4096] f32
    float* out = (float*)d_out;               // [16384] f32 weights + [16384] i32 indices

    const dim3 grid(NUM_TOKENS / BM);
    const dim3 block(THREADS);
    constexpr size_t WBF_BYTES = (size_t)N_EXPERTS * DIM * 2;   // 2 MB bf16 weight

    if (ws_size >= WBF_BYTES) {
        unsigned int* wb = (unsigned int*)d_ws;
        cvt_weight_kernel<<<(N_EXPERTS * DIM / 2) / 256, 256, 0, stream>>>(w, wb);
        hipFuncSetAttribute(reinterpret_cast<const void*>(router_kernel<true>),
                            hipFuncAttributeMaxDynamicSharedMemorySize,
                            (int)SMEM_BYTES);
        router_kernel<true><<<grid, block, SMEM_BYTES, stream>>>(
            x, w, (const unsigned short*)wb, out);
    } else {
        hipFuncSetAttribute(reinterpret_cast<const void*>(router_kernel<false>),
                            hipFuncAttributeMaxDynamicSharedMemorySize,
                            (int)SMEM_BYTES);
        router_kernel<false><<<grid, block, SMEM_BYTES, stream>>>(
            x, w, nullptr, out);
    }
}
